// LlamaMoELayer_55138790146429
// MI455X (gfx1250) — matrix operand order repaired and checked by reading, not compiled, not run on hardware
//
#include <hip/hip_runtime.h>
#include <hip/hip_bf16.h>

// MoE: T=8192 tokens, H=2048, I=5632, E=8, top-2 (dense compute, combine-scaled).
#define T_TOK 8192
#define H_DIM 2048
#define I_DIM 5632
#define E_NUM 8
#define K_TOP 2

typedef __attribute__((ext_vector_type(16))) __bf16 v16bf;
typedef __attribute__((ext_vector_type(8)))  float  v8f;

union BV { unsigned int u[8]; v16bf v; };

__device__ __forceinline__ unsigned short f2bf(float f) {
  unsigned int u = __float_as_uint(f);
  unsigned int r = u + 0x7FFFu + ((u >> 16) & 1u);   // round-to-nearest-even
  return (unsigned short)(r >> 16);
}
__device__ __forceinline__ unsigned int pack2bf(float a, float b) {
#if __has_builtin(__builtin_amdgcn_cvt_pk_bf16_f32)
  typedef __attribute__((ext_vector_type(2))) __bf16 v2bf;
  union { v2bf v; unsigned int u; } cv;
  cv.v = __builtin_amdgcn_cvt_pk_bf16_f32(a, b);
  return cv.u;
#else
  return (unsigned int)f2bf(a) | ((unsigned int)f2bf(b) << 16);
#endif
}

// A-matrix fragment (16x32 bf16), CDNA5 layout (ISA 7.12.2):
// lane m=l%16; VGPR v holds k-pair p=(v&3)+(l>=16?4:0)+(v>=4?8:0)
__device__ __forceinline__ v16bf fetchA(const unsigned int* __restrict__ lds,
                                        int row0, int lane) {
  BV r;
  const int m  = row0 + (lane & 15);
  const int hi = (lane >> 4) & 1;
#pragma unroll
  for (int v = 0; v < 8; ++v) {
    const int p = (v & 3) + (hi ? 4 : 0) + ((v >> 2) ? 8 : 0);
    r.u[v] = lds[m * 17 + p];
  }
  return r.v;
}
// B-matrix fragment (32x16 bf16): same k-pair order as A, p=(v&3)+(l>=16?4:0)+(v>=4?8:0).
__device__ __forceinline__ v16bf fetchB(const unsigned int* __restrict__ lds,
                                        int col0, int lane) {
  BV r;
  const int n  = col0 + (lane & 15);
  const int hi = (lane >> 4) & 1;
#pragma unroll
  for (int v = 0; v < 8; ++v) {
    const int p = (v & 3) + (hi ? 4 : 0) + ((v >> 2) ? 8 : 0);
    r.u[v] = lds[n * 17 + p];
  }
  return r.v;
}

// Async verbatim copy of a 64x32 bf16 tile (row-major, rowStride elements)
// into padded LDS [64][17] dwords, via GLOBAL_LOAD_ASYNC_TO_LDS (ASYNCcnt).
// 1024 dwords / 256 threads = 4 per thread; b32 keeps any alignment legal.
__device__ __forceinline__ void asyncTileA(unsigned int* ldsA,
                                           const unsigned short* src,
                                           size_t rowStride, int m0, int k0,
                                           int tid) {
#pragma unroll
  for (int j = 0; j < 4; ++j) {
    const int p = tid + j * 256;
    const int m = p >> 4, kp = p & 15;
    const unsigned int* gp =
        (const unsigned int*)(src + (size_t)(m0 + m) * rowStride + k0) + kp;
    const unsigned int lv =
        (unsigned int)(unsigned long long)(&ldsA[m * 17 + kp]);
    asm volatile("global_load_async_to_lds_b32 %0, %1, off"
                 :: "v"(lv), "v"(gp) : "memory");
  }
}
__device__ __forceinline__ void waitAsync() {
  asm volatile("s_wait_asynccnt 0" ::: "memory");
}

// ---------------- Router: one wave32 per token (+ x -> bf16 cast) ----------------
__global__ __launch_bounds__(256) void moe_router_kernel(
    const float* __restrict__ x, const float* __restrict__ rw,
    float* __restrict__ logits, float* __restrict__ topi,
    float* __restrict__ topw, float* __restrict__ combine,
    unsigned short* __restrict__ xb, int useXb) {
  const int gtid = blockIdx.x * blockDim.x + threadIdx.x;
  const int t    = gtid >> 5;
  const int lane = gtid & 31;
  if (t >= T_TOK) return;
  const float* xt = x + (size_t)t * H_DIM;
  float acc[E_NUM];
#pragma unroll
  for (int e = 0; e < E_NUM; ++e) acc[e] = 0.0f;
  for (int h = lane; h < H_DIM; h += 32) {
    const float xv = xt[h];
    if (useXb) xb[(size_t)t * H_DIM + h] = f2bf(xv);
#pragma unroll
    for (int e = 0; e < E_NUM; ++e) acc[e] += xv * rw[e * H_DIM + h];
  }
#pragma unroll
  for (int e = 0; e < E_NUM; ++e)
#pragma unroll
    for (int off = 16; off > 0; off >>= 1) acc[e] += __shfl_xor(acc[e], off, 32);

  if (lane == 0) {
    float mx = acc[0];
#pragma unroll
    for (int e = 1; e < E_NUM; ++e) mx = fmaxf(mx, acc[e]);
    float p[E_NUM], s = 0.0f;
#pragma unroll
    for (int e = 0; e < E_NUM; ++e) { p[e] = __expf(acc[e] - mx); s += p[e]; }
    const float inv = __builtin_amdgcn_rcpf(s);
#pragma unroll
    for (int e = 0; e < E_NUM; ++e) p[e] *= inv;
    int i0 = 0;
#pragma unroll
    for (int e = 1; e < E_NUM; ++e) if (p[e] > p[i0]) i0 = e;
    int i1 = (i0 == 0) ? 1 : 0;
#pragma unroll
    for (int e = 0; e < E_NUM; ++e) if (e != i0 && p[e] > p[i1]) i1 = e;
    float w0 = p[i0], w1 = p[i1];
    const float wn = __builtin_amdgcn_rcpf(w0 + w1);
    w0 *= wn; w1 *= wn;
#pragma unroll
    for (int e = 0; e < E_NUM; ++e) {
      logits[(size_t)t * E_NUM + e]  = acc[e];
      combine[(size_t)t * E_NUM + e] = 0.0f;
    }
    combine[(size_t)t * E_NUM + i0] = w0;
    combine[(size_t)t * E_NUM + i1] = w1;
    topi[t * K_TOP + 0] = (float)i0;
    topi[t * K_TOP + 1] = (float)i1;
    topw[t * K_TOP + 0] = w0;
    topw[t * K_TOP + 1] = w1;
  }
}

// ------- Fused gate+up GEMM + SiLU: act = silu(x@Wg) * (x@Wu), bf16 out -------
// Block tile 64(M) x 128(N), 8 waves each own 32x32. K = H in steps of 32.
// ASYNC_A=1: A-tile comes from pre-cast bf16 xb via async-to-LDS DMA.
template <int ASYNC_A>
__global__ __launch_bounds__(256) void moe_gateup_kernel(
    const float* __restrict__ x,          // [T,H] f32
    const unsigned short* __restrict__ xb,// [T,H] bf16 bits (ASYNC_A only)
    const float* __restrict__ wg,         // [H,I] for this expert
    const float* __restrict__ wu,         // [H,I]
    unsigned short* __restrict__ act,     // [Tc,I] bf16 bits
    int t0) {
  __shared__ unsigned int ldsA[64 * 17];
  __shared__ unsigned int ldsG[128 * 17];
  __shared__ unsigned int ldsU[128 * 17];

  const int tid  = threadIdx.x;
  const int lane = tid & 31;
  const int w    = tid >> 5;
  const int wm   = w & 1;       // 0..1 -> M offset
  const int wn   = w >> 1;      // 0..3 -> N offset
  const int n0   = blockIdx.x * 128;
  const int m0   = blockIdx.y * 64;     // chunk-local token offset

  const v8f vz = {};
  v8f accG[2][2], accU[2][2];
#pragma unroll
  for (int mi = 0; mi < 2; ++mi)
#pragma unroll
    for (int ni = 0; ni < 2; ++ni) { accG[mi][ni] = vz; accU[mi][ni] = vz; }

  for (int k0 = 0; k0 < H_DIM; k0 += 32) {
    // A tile: 64x32 tokens x H-chunk
    if (ASYNC_A) {
      asyncTileA(ldsA, xb, H_DIM, t0 + m0, k0, tid);
    } else {
#pragma unroll
      for (int j = 0; j < 4; ++j) {
        const int p = tid + j * 256;           // 1024 dword-pairs
        const int m = p >> 4, kp = p & 15;
        const float* src = x + (size_t)(t0 + m0 + m) * H_DIM + k0 + 2 * kp;
        ldsA[m * 17 + kp] = pack2bf(src[0], src[1]);
      }
    }
    // B tiles: Wg/Wu rows k0..k0+31, cols n0..n0+127, stored [n][kpair]
#pragma unroll
    for (int j = 0; j < 8; ++j) {
      const int p = tid + j * 256;           // 2048 dword-pairs
      const int n = p & 127, kp = p >> 7;
      const size_t o0 = (size_t)(k0 + 2 * kp) * I_DIM + n0 + n;
      ldsG[n * 17 + kp] = pack2bf(wg[o0], wg[o0 + I_DIM]);
      ldsU[n * 17 + kp] = pack2bf(wu[o0], wu[o0 + I_DIM]);
    }
    if (k0 + 32 < H_DIM) {  // prefetch next K-chunk of weights toward L2
      const size_t pn = (size_t)(k0 + 32 + (tid >> 7)) * I_DIM + n0 + (tid & 127);
      __builtin_prefetch(&wg[pn], 0, 1);
      __builtin_prefetch(&wu[pn], 0, 1);
    }
    if (ASYNC_A) waitAsync();
    __syncthreads();

    v16bf a[2], bg[2], bu[2];
#pragma unroll
    for (int mi = 0; mi < 2; ++mi) a[mi]  = fetchA(ldsA, wm * 32 + 16 * mi, lane);
#pragma unroll
    for (int ni = 0; ni < 2; ++ni) { bg[ni] = fetchB(ldsG, wn * 32 + 16 * ni, lane);
                                     bu[ni] = fetchB(ldsU, wn * 32 + 16 * ni, lane); }
#pragma unroll
    for (int mi = 0; mi < 2; ++mi)
#pragma unroll
      for (int ni = 0; ni < 2; ++ni) {
        accG[mi][ni] = __builtin_amdgcn_wmma_f32_16x16x32_bf16(
            false, a[mi], false, bg[ni], (short)0, accG[mi][ni], false, false);
        accU[mi][ni] = __builtin_amdgcn_wmma_f32_16x16x32_bf16(
            false, a[mi], false, bu[ni], (short)0, accU[mi][ni], false, false);
      }
    __syncthreads();
  }

  // Epilogue: silu(g)*u -> bf16 act. C layout: m = r + (lane>=16?8:0), n = lane%16.
  const int hi8 = ((lane >> 4) & 1) * 8;
#pragma unroll
  for (int mi = 0; mi < 2; ++mi)
#pragma unroll
    for (int ni = 0; ni < 2; ++ni) {
      const int nC = n0 + wn * 32 + ni * 16 + (lane & 15);
#pragma unroll
      for (int r = 0; r < 8; ++r) {
        const int m = m0 + wm * 32 + mi * 16 + r + hi8;
        const float g = accG[mi][ni][r];
        const float u = accU[mi][ni][r];
        const float s = g * __builtin_amdgcn_rcpf(1.0f + __expf(-g)) * u;
        act[(size_t)m * I_DIM + nC] = f2bf(s);
      }
    }
}

// ------- Down GEMM: out[t,:] (+)= combine[t,e] * (act @ Wd) -------
__global__ __launch_bounds__(256) void moe_down_kernel(
    const unsigned short* __restrict__ act, // [Tc,I] bf16 bits
    const float* __restrict__ wd,           // [I,H] this expert
    const float* __restrict__ combine,      // [T,E]
    float* __restrict__ out,                // [T,H]
    int t0, int e, int beta) {
  __shared__ unsigned int ldsA[64 * 17];
  __shared__ unsigned int ldsB[128 * 17];

  const int tid  = threadIdx.x;
  const int lane = tid & 31;
  const int w    = tid >> 5;
  const int wm   = w & 1;
  const int wn   = w >> 1;
  const int n0   = blockIdx.x * 128;
  const int m0   = blockIdx.y * 64;

  const v8f vz = {};
  v8f acc[2][2];
#pragma unroll
  for (int mi = 0; mi < 2; ++mi)
#pragma unroll
    for (int ni = 0; ni < 2; ++ni) acc[mi][ni] = vz;

  for (int k0 = 0; k0 < I_DIM; k0 += 32) {
    // A tile: act is already bf16 -> async DMA straight into LDS
    asyncTileA(ldsA, act, I_DIM, m0, k0, tid);
#pragma unroll
    for (int j = 0; j < 8; ++j) {
      const int p = tid + j * 256;
      const int n = p & 127, kp = p >> 7;
      const size_t o0 = (size_t)(k0 + 2 * kp) * H_DIM + n0 + n;
      ldsB[n * 17 + kp] = pack2bf(wd[o0], wd[o0 + H_DIM]);
    }
    if (k0 + 32 < I_DIM) {
      const size_t pn = (size_t)(k0 + 32 + (tid >> 7)) * H_DIM + n0 + (tid & 127);
      __builtin_prefetch(&wd[pn], 0, 1);
    }
    waitAsync();
    __syncthreads();

    v16bf a[2], b[2];
#pragma unroll
    for (int mi = 0; mi < 2; ++mi) a[mi] = fetchA(ldsA, wm * 32 + 16 * mi, lane);
#pragma unroll
    for (int ni = 0; ni < 2; ++ni) b[ni] = fetchB(ldsB, wn * 32 + 16 * ni, lane);
#pragma unroll
    for (int mi = 0; mi < 2; ++mi)
#pragma unroll
      for (int ni = 0; ni < 2; ++ni)
        acc[mi][ni] = __builtin_amdgcn_wmma_f32_16x16x32_bf16(
            false, a[mi], false, b[ni], (short)0, acc[mi][ni], false, false);
    __syncthreads();
  }

  const int hi8 = ((lane >> 4) & 1) * 8;
#pragma unroll
  for (int mi = 0; mi < 2; ++mi)
#pragma unroll
    for (int ni = 0; ni < 2; ++ni) {
      const int nC = n0 + wn * 32 + ni * 16 + (lane & 15);
#pragma unroll
      for (int r = 0; r < 8; ++r) {
        const int m   = m0 + wm * 32 + mi * 16 + r + hi8;
        const int tok = t0 + m;
        const float cw  = combine[(size_t)tok * E_NUM + e];
        const float val = acc[mi][ni][r] * cw;
        const size_t oi = (size_t)tok * H_DIM + nC;
        out[oi] = beta ? (out[oi] + val) : val;
      }
    }
}

extern "C" void kernel_launch(void* const* d_in, const int* in_sizes, int n_in,
                              void* d_out, int out_size, void* d_ws, size_t ws_size,
                              hipStream_t stream) {
  const float* x  = (const float*)d_in[0];  // [B,S,H]
  const float* rw = (const float*)d_in[1];  // [E,H]
  const float* wg = (const float*)d_in[2];  // [E,H,I]
  const float* wu = (const float*)d_in[3];  // [E,H,I]
  const float* wd = (const float*)d_in[4];  // [E,I,H]

  float* out    = (float*)d_out;                       // [T,H]
  float* logits = out + (size_t)T_TOK * H_DIM;         // [T,E]
  float* topi   = logits + (size_t)T_TOK * E_NUM;      // [T,K] (as float)
  float* topw   = topi + (size_t)T_TOK * K_TOP;        // [T,K]

  float* combine = (float*)d_ws;                       // [T,E]
  size_t off = (((size_t)T_TOK * E_NUM * sizeof(float)) + 255) & ~(size_t)255;

  const size_t xbBytes  = (size_t)T_TOK * H_DIM * sizeof(unsigned short);
  const size_t minActBy = (size_t)64 * I_DIM * sizeof(unsigned short);
  int useXb = (ws_size >= off + xbBytes + minActBy) ? 1 : 0;
  unsigned short* xb = (unsigned short*)((char*)d_ws + off);
  if (useXb) off += xbBytes;
  unsigned short* act = (unsigned short*)((char*)d_ws + off);

  // Token chunk size so act[Tc,I] bf16 fits in the remaining workspace.
  int Tc = T_TOK;
  while (Tc > 64 &&
         off + (size_t)Tc * I_DIM * sizeof(unsigned short) > ws_size)
    Tc >>= 1;

  moe_router_kernel<<<T_TOK / 8, 256, 0, stream>>>(x, rw, logits, topi, topw,
                                                   combine, xb, useXb);

  for (int t0 = 0; t0 < T_TOK; t0 += Tc) {
    for (int e = 0; e < E_NUM; ++e) {
      if (useXb)
        moe_gateup_kernel<1><<<dim3(I_DIM / 128, Tc / 64), 256, 0, stream>>>(
            x, xb, wg + (size_t)e * H_DIM * I_DIM,
            wu + (size_t)e * H_DIM * I_DIM, act, t0);
      else
        moe_gateup_kernel<0><<<dim3(I_DIM / 128, Tc / 64), 256, 0, stream>>>(
            x, xb, wg + (size_t)e * H_DIM * I_DIM,
            wu + (size_t)e * H_DIM * I_DIM, act, t0);
      moe_down_kernel<<<dim3(H_DIM / 128, Tc / 64), 256, 0, stream>>>(
          act, wd + (size_t)e * I_DIM * H_DIM, combine, out, t0, e,
          e > 0 ? 1 : 0);
    }
  }
}
